// SENNGC_attention_per_lag_2405181686005
// MI455X (gfx1250) — matrix operand order repaired and checked by reading, not compiled, not run on hardware
//
#include <hip/hip_runtime.h>
#include <hip/hip_bf16.h>
#include <math.h>

// Problem constants (match reference)
#define Pn 128
#define Kn 10
#define Hn 512
#define Bn 1024
#define PPn (Pn * Pn)   // 16384
#define Ln1 2           // L-1 hidden layers

typedef __bf16 bf16_t;
typedef __attribute__((ext_vector_type(16))) __bf16 v16bf;
typedef __attribute__((ext_vector_type(8)))  __bf16 v8bf;
typedef __attribute__((ext_vector_type(8)))  float  v8f;

// ---------------------------------------------------------------------------
// Kernel 1: per-(b,k) gated window summary -> bf16 summary [B,K,P]
// ---------------------------------------------------------------------------
__global__ __launch_bounds__(Pn)
void gate_kernel(const float* __restrict__ inputs,  // [B,K,P]
                 const float* __restrict__ Wg,      // [K, 3P, 3]
                 const float* __restrict__ bg,      // [K, 3]
                 bf16_t* __restrict__ summary)      // [B,K,P] bf16
{
    const int bk = blockIdx.x;
    const int b  = bk / Kn;
    const int k  = bk % Kn;
    const int p  = threadIdx.x;

    // windows[w] = inputs[b, k-1+w, p] with zero pad
    float x0 = (k >= 1)      ? inputs[((size_t)b * Kn + (k - 1)) * Pn + p] : 0.f;
    float x1 =                 inputs[((size_t)b * Kn + k)       * Pn + p];
    float x2 = (k + 1 < Kn)  ? inputs[((size_t)b * Kn + (k + 1)) * Pn + p] : 0.f;

    // partial logits: sum over f = w_in*P + p of flat[f] * Wg[k][f][w]
    float pl0, pl1, pl2;
    const float* wgk = Wg + (size_t)k * (3 * Pn) * 3;
    {
        const float* g0 = wgk + (size_t)(0 * Pn + p) * 3;
        const float* g1 = wgk + (size_t)(1 * Pn + p) * 3;
        const float* g2 = wgk + (size_t)(2 * Pn + p) * 3;
        pl0 = x0 * g0[0] + x1 * g1[0] + x2 * g2[0];
        pl1 = x0 * g0[1] + x1 * g1[1] + x2 * g2[1];
        pl2 = x0 * g0[2] + x1 * g1[2] + x2 * g2[2];
    }

    __shared__ float lred[3][Pn];
    lred[0][p] = pl0; lred[1][p] = pl1; lred[2][p] = pl2;
    __syncthreads();
    for (int s = Pn / 2; s > 0; s >>= 1) {
        if (p < s) {
            lred[0][p] += lred[0][p + s];
            lred[1][p] += lred[1][p + s];
            lred[2][p] += lred[2][p + s];
        }
        __syncthreads();
    }
    float l0 = lred[0][0] + bg[k * 3 + 0];
    float l1 = lred[1][0] + bg[k * 3 + 1];
    float l2 = lred[2][0] + bg[k * 3 + 2];
    float m  = fmaxf(l0, fmaxf(l1, l2));
    float e0 = __expf(l0 - m), e1 = __expf(l1 - m), e2 = __expf(l2 - m);
    float inv = 1.f / (e0 + e1 + e2);
    float s0 = e0 * inv, s1 = e1 * inv, s2 = e2 * inv;

    float sumv = s0 * x0 + s1 * x1 + s2 * x2;
    summary[((size_t)b * Kn + k) * Pn + p] = (bf16_t)sumv;
}

// ---------------------------------------------------------------------------
// Kernel 2: batched f32 [R,C] -> bf16 transposed [C,R] (LDS-tiled)
// grid: (C/32, R/32, batch), block: (32, 8)
// ---------------------------------------------------------------------------
__global__ __launch_bounds__(256)
void transpose_convert(const float* __restrict__ src, bf16_t* __restrict__ dst,
                       int R, int C, long long src_bstride, long long dst_bstride)
{
    __shared__ float tile[32][33];
    const int c0 = blockIdx.x * 32;
    const int r0 = blockIdx.y * 32;
    src += (long long)blockIdx.z * src_bstride;
    dst += (long long)blockIdx.z * dst_bstride;
    const int tx = threadIdx.x;
    #pragma unroll
    for (int i = threadIdx.y; i < 32; i += 8)
        tile[i][tx] = src[(long long)(r0 + i) * C + (c0 + tx)];
    __syncthreads();
    #pragma unroll
    for (int i = threadIdx.y; i < 32; i += 8)
        dst[(long long)(c0 + i) * R + (r0 + tx)] = (bf16_t)tile[tx][i];
}

// ---------------------------------------------------------------------------
// WMMA fragment helpers (ISA bf16 layouts, wave32)
//   A 16x32: lane half 0 -> K chunks [0..7],[16..23]; half 1 -> [8..15],[24..31]
//   B 32x16: lane half 0 -> K chunks [0..7],[16..23]; half 1 -> [8..15],[24..31] of column n
// ---------------------------------------------------------------------------
__device__ __forceinline__ v16bf load_frag(const bf16_t* p, int off0, int off1)
{
    v8bf lo = *(const v8bf*)(p + off0);
    v8bf hi = *(const v8bf*)(p + off1);
    v16bf f;
    #pragma unroll
    for (int i = 0; i < 8; ++i) { f[i] = lo[i]; f[8 + i] = hi[i]; }
    return f;
}

template <int ACT, typename OutT>
__device__ __forceinline__ void store_subtile(OutT* __restrict__ op, const v8f& c,
                                              float bv, long long out_m_stride,
                                              int row_base, long long col)
{
    #pragma unroll
    for (int r = 0; r < 8; ++r) {
        float v = c[r] + bv;
        if (ACT == 1) v = fmaxf(v, 0.f);
        else if (ACT == 2) v = tanhf(v);
        op[(long long)(row_base + r) * out_m_stride + col] = (OutT)v;
    }
}

// ---------------------------------------------------------------------------
// Kernel 3: batched WMMA GEMM  out = act(A @ Bt^T + bias)
//   A  : bf16, element (m,kd,batch) at A[m*A_m_stride + batch*A_b_stride + kd]
//   Bt : bf16, pre-transposed: Bt[batch*B_b_stride + n*Kd + kd]
//   out: element (m,n,batch) at out[m*out_m_stride + batch*out_b_stride + n]
//   ACT: 0=none, 1=relu, 2=tanh
// Register-blocked: each wave computes 64x32 (4 M-subtiles x 2 N-subtiles),
// 8 WMMAs per 32-deep K-step on 6 fragment loads. Block = 8 waves = 128x128.
// grid: (N/128, M/128, nbatch)
// ---------------------------------------------------------------------------
template <int ACT, typename OutT>
__global__ __launch_bounds__(256)
void wmma_gemm_bias_act(const bf16_t* __restrict__ A,
                        const bf16_t* __restrict__ Bt,
                        const float*  __restrict__ bias,
                        OutT* __restrict__ out,
                        int Kd,
                        long long A_m_stride, long long A_b_stride,
                        long long B_b_stride, long long bias_b_stride,
                        long long out_m_stride, long long out_b_stride)
{
    const int lane = threadIdx.x & 31;
    const int wave = threadIdx.x >> 5;
    const int wm   = wave >> 2;        // 0..1 -> 64-row group
    const int wn   = wave & 3;         // 0..3 -> 32-col group
    const int half = lane >> 4;
    const int l16  = lane & 15;

    const int m0    = blockIdx.y * 128 + wm * 64;
    const int n0    = blockIdx.x * 128 + wn * 32;
    const int batch = blockIdx.z;

    const bf16_t* a0 = A + (long long)(m0 +  0 + l16) * A_m_stride + (long long)batch * A_b_stride;
    const bf16_t* a1 = A + (long long)(m0 + 16 + l16) * A_m_stride + (long long)batch * A_b_stride;
    const bf16_t* a2 = A + (long long)(m0 + 32 + l16) * A_m_stride + (long long)batch * A_b_stride;
    const bf16_t* a3 = A + (long long)(m0 + 48 + l16) * A_m_stride + (long long)batch * A_b_stride;
    const bf16_t* b0 = Bt + (long long)batch * B_b_stride + (long long)(n0 +  0 + l16) * Kd;
    const bf16_t* b1 = Bt + (long long)batch * B_b_stride + (long long)(n0 + 16 + l16) * Kd;

    const int aoff0 = half * 8;        // first A chunk within K-step
    const int aoff1 = 16 + half * 8;   // second A chunk
    const int boff  = half * 8;        // B chunks (same order as A)

    v8f c00 = {}, c01 = {}, c10 = {}, c11 = {};
    v8f c20 = {}, c21 = {}, c30 = {}, c31 = {};

    for (int kk = 0; kk < Kd; kk += 32) {
        if (kk + 64 < Kd) {            // keep the L2->L0 pipe primed for B
            __builtin_prefetch(b0 + kk + 64, 0, 1);
            __builtin_prefetch(b1 + kk + 64, 0, 1);
        }
        v16bf fb0 = load_frag(b0 + kk, boff, boff + 16);
        v16bf fb1 = load_frag(b1 + kk, boff, boff + 16);
        v16bf fa0 = load_frag(a0 + kk, aoff0, aoff1);
        v16bf fa1 = load_frag(a1 + kk, aoff0, aoff1);
        v16bf fa2 = load_frag(a2 + kk, aoff0, aoff1);
        v16bf fa3 = load_frag(a3 + kk, aoff0, aoff1);

        c00 = __builtin_amdgcn_wmma_f32_16x16x32_bf16(false, fa0, false, fb0, (short)0, c00, false, false);
        c01 = __builtin_amdgcn_wmma_f32_16x16x32_bf16(false, fa0, false, fb1, (short)0, c01, false, false);
        c10 = __builtin_amdgcn_wmma_f32_16x16x32_bf16(false, fa1, false, fb0, (short)0, c10, false, false);
        c11 = __builtin_amdgcn_wmma_f32_16x16x32_bf16(false, fa1, false, fb1, (short)0, c11, false, false);
        c20 = __builtin_amdgcn_wmma_f32_16x16x32_bf16(false, fa2, false, fb0, (short)0, c20, false, false);
        c21 = __builtin_amdgcn_wmma_f32_16x16x32_bf16(false, fa2, false, fb1, (short)0, c21, false, false);
        c30 = __builtin_amdgcn_wmma_f32_16x16x32_bf16(false, fa3, false, fb0, (short)0, c30, false, false);
        c31 = __builtin_amdgcn_wmma_f32_16x16x32_bf16(false, fa3, false, fb1, (short)0, c31, false, false);
    }

    const float bv0 = bias[(long long)batch * bias_b_stride + (n0 + l16)];
    const float bv1 = bias[(long long)batch * bias_b_stride + (n0 + 16 + l16)];
    OutT* op = out + (long long)batch * out_b_stride;
    const long long col0 = n0 + l16;
    const long long col1 = n0 + 16 + l16;
    const int rb = half * 8;   // D layout: VGPR r -> row half*8 + r

    store_subtile<ACT, OutT>(op, c00, bv0, out_m_stride, m0 +  0 + rb, col0);
    store_subtile<ACT, OutT>(op, c01, bv1, out_m_stride, m0 +  0 + rb, col1);
    store_subtile<ACT, OutT>(op, c10, bv0, out_m_stride, m0 + 16 + rb, col0);
    store_subtile<ACT, OutT>(op, c11, bv1, out_m_stride, m0 + 16 + rb, col1);
    store_subtile<ACT, OutT>(op, c20, bv0, out_m_stride, m0 + 32 + rb, col0);
    store_subtile<ACT, OutT>(op, c21, bv1, out_m_stride, m0 + 32 + rb, col1);
    store_subtile<ACT, OutT>(op, c30, bv0, out_m_stride, m0 + 48 + rb, col0);
    store_subtile<ACT, OutT>(op, c31, bv1, out_m_stride, m0 + 48 + rb, col1);
}

// ---------------------------------------------------------------------------
// Kernel 4: preds[b,i] = sum_k sum_j coeffs[b,k,i,j] * inputs[b,k,j]
// grid: B blocks, 128 threads
// ---------------------------------------------------------------------------
__global__ __launch_bounds__(Pn)
void preds_kernel(const float* __restrict__ coeffs,  // [B,K,P,P]
                  const float* __restrict__ inputs,  // [B,K,P]
                  float* __restrict__ preds)         // [B,P]
{
    const int b = blockIdx.x;
    const int i = threadIdx.x;
    __shared__ float xs[Pn];
    float acc = 0.f;
    for (int k = 0; k < Kn; ++k) {
        xs[i] = inputs[((size_t)b * Kn + k) * Pn + i];
        __syncthreads();
        const float* crow = coeffs + (((size_t)b * Kn + k) * Pn + i) * Pn;
        #pragma unroll 8
        for (int j = 0; j < Pn; ++j) acc += crow[j] * xs[j];
        __syncthreads();
    }
    preds[(size_t)b * Pn + i] = acc;
}

// ---------------------------------------------------------------------------
extern "C" void kernel_launch(void* const* d_in, const int* in_sizes, int n_in,
                              void* d_out, int out_size, void* d_ws, size_t ws_size,
                              hipStream_t stream) {
    (void)in_sizes; (void)n_in; (void)out_size; (void)ws_size;

    const float* inputs = (const float*)d_in[0];   // [B,K,P]
    const float* Wg     = (const float*)d_in[1];   // [K,3P,3]
    const float* bg     = (const float*)d_in[2];   // [K,3]
    const float* W1     = (const float*)d_in[3];   // [K,P,H]
    const float* b1     = (const float*)d_in[4];   // [K,H]
    const float* Wh     = (const float*)d_in[5];   // [2,K,H,H]
    const float* bh     = (const float*)d_in[6];   // [2,K,H]
    const float* Wout   = (const float*)d_in[7];   // [K,H,P*P]
    const float* bout   = (const float*)d_in[8];   // [K,P*P]

    float* preds  = (float*)d_out;                     // [B,P]
    float* coeffs = preds + (size_t)Bn * Pn;           // [B,K,P,P]

    // Workspace layout (bf16), total ~203 MB
    bf16_t* summaryB = (bf16_t*)d_ws;                            // B*K*P
    bf16_t* W1T      = summaryB + (size_t)Bn * Kn * Pn;          // K*H*P
    bf16_t* WhT      = W1T      + (size_t)Kn * Hn * Pn;          // 2*K*H*H
    bf16_t* WoutT    = WhT      + (size_t)Ln1 * Kn * Hn * Hn;    // K*PP*H
    bf16_t* h0       = WoutT    + (size_t)Kn * PPn * Hn;         // B*K*H
    bf16_t* h1       = h0       + (size_t)Bn * Kn * Hn;          // B*K*H

    // 1) Gate + summary
    gate_kernel<<<Bn * Kn, Pn, 0, stream>>>(inputs, Wg, bg, summaryB);

    // 2) Weight conversions (f32 -> bf16, transposed so B-columns are contiguous)
    transpose_convert<<<dim3(Hn / 32, Pn / 32, Kn), dim3(32, 8), 0, stream>>>(
        W1, W1T, Pn, Hn, (long long)Pn * Hn, (long long)Hn * Pn);
    transpose_convert<<<dim3(Hn / 32, Hn / 32, Ln1 * Kn), dim3(32, 8), 0, stream>>>(
        Wh, WhT, Hn, Hn, (long long)Hn * Hn, (long long)Hn * Hn);
    transpose_convert<<<dim3(PPn / 32, Hn / 32, Kn), dim3(32, 8), 0, stream>>>(
        Wout, WoutT, Hn, PPn, (long long)Hn * PPn, (long long)PPn * Hn);

    // 3) h0 = relu(summary @ W1 + b1)   [M=B=1024, N=H, Kd=P]
    wmma_gemm_bias_act<1, bf16_t><<<dim3(Hn / 128, Bn / 128, Kn), 256, 0, stream>>>(
        summaryB, W1T, b1, h0, Pn,
        (long long)Kn * Pn, (long long)Pn,
        (long long)Hn * Pn, (long long)Hn,
        (long long)Kn * Hn, (long long)Hn);

    // 4) h1 = relu(h0 @ Wh[0] + bh[0])  [Kd=H]
    wmma_gemm_bias_act<1, bf16_t><<<dim3(Hn / 128, Bn / 128, Kn), 256, 0, stream>>>(
        h0, WhT, bh, h1, Hn,
        (long long)Kn * Hn, (long long)Hn,
        (long long)Hn * Hn, (long long)Hn,
        (long long)Kn * Hn, (long long)Hn);

    // 5) h0 = relu(h1 @ Wh[1] + bh[1])
    wmma_gemm_bias_act<1, bf16_t><<<dim3(Hn / 128, Bn / 128, Kn), 256, 0, stream>>>(
        h1, WhT + (size_t)Kn * Hn * Hn, bh + (size_t)Kn * Hn, h0, Hn,
        (long long)Kn * Hn, (long long)Hn,
        (long long)Hn * Hn, (long long)Hn,
        (long long)Kn * Hn, (long long)Hn);

    // 6) coeffs = tanh(h0 @ Wout + bout)  [N=P*P=16384], f32 straight into d_out
    wmma_gemm_bias_act<2, float><<<dim3(PPn / 128, Bn / 128, Kn), 256, 0, stream>>>(
        h0, WoutT, bout, coeffs, Hn,
        (long long)Kn * Hn, (long long)Hn,
        (long long)PPn * Hn, (long long)PPn,
        (long long)Kn * PPn, (long long)PPn);

    // 7) preds = sum_k coeffs_k @ x_k
    preds_kernel<<<Bn, Pn, 0, stream>>>(coeffs, inputs, preds);
}
